// SelfAttention_58007828299801
// MI455X (gfx1250) — matrix operand order repaired and checked by reading, not compiled, not run on hardware
//
#include <hip/hip_runtime.h>
#include <stdint.h>

// ---------------------------------------------------------------------------
// Self-attention, MI455X (gfx1250, wave32, WMMA).
//   B=4, L=2048, E=1024, H=16, D=64.
//   Pipeline: 3x GEMM (fp32 in -> bf16 out; Q pre-scaled by 1/sqrt(E);
//                      V stored transposed per head)
//             -> flash attention (bf16 WMMA, fp32 softmax/accum,
//                row-max via single-instruction DPP butterfly,
//                row-sum via WMMA-with-ones)
//             -> output GEMM (bf16 in via async-to-LDS copy -> fp32 out).
// ---------------------------------------------------------------------------

typedef __bf16  v16bf __attribute__((ext_vector_type(16)));
typedef __bf16  bf8   __attribute__((ext_vector_type(8)));
typedef float   v8f   __attribute__((ext_vector_type(8)));

union uv16 { v16bf v; bf8 h[2]; };

__device__ __forceinline__ v8f v8f_zero() {
    v8f z;
#pragma unroll
    for (int i = 0; i < 8; ++i) z[i] = 0.0f;
    return z;
}

__device__ __forceinline__ v8f wmma_bf16(v16bf a, v16bf b, v8f c) {
    return __builtin_amdgcn_wmma_f32_16x16x32_bf16(
        /*neg_a=*/false, a, /*neg_b=*/false, b,
        /*c_mod=*/(short)0, c, /*reuse_a=*/false, /*reuse_b=*/false);
}

// Full 16-lane-row max via DPP butterfly, one v_max_num_f32_dpp per stage:
// xor1 (quad_perm 1,0,3,2), xor2 (quad_perm 2,3,0,1),
// xor7 (row_half_mirror), xor15 (row_mirror).
__device__ __forceinline__ float red_max16(float v) {
    float r;
    asm("v_max_num_f32_dpp %0, %1, %1 quad_perm:[1,0,3,2] row_mask:0xf bank_mask:0xf"
        : "=v"(r) : "v"(v));
    asm("v_max_num_f32_dpp %0, %1, %1 quad_perm:[2,3,0,1] row_mask:0xf bank_mask:0xf"
        : "=v"(v) : "v"(r));
    asm("v_max_num_f32_dpp %0, %1, %1 row_half_mirror row_mask:0xf bank_mask:0xf"
        : "=v"(r) : "v"(v));
    asm("v_max_num_f32_dpp %0, %1, %1 row_mirror row_mask:0xf bank_mask:0xf"
        : "=v"(v) : "v"(r));
    return v;
}

// ---------------------------------------------------------------------------
// GEMM:  Out[m][n] = (sum_k A[m][k] * W[n][k] + bias[n]) * oscale
//   A: M x K row-major (fp32 or bf16), W: N x K row-major fp32.
//   MODE 0: store bf16 row-major [M][N]
//   MODE 1: store bf16 transposed-per-head: [b][h][d][L]  (for V)
//   MODE 2: store fp32 row-major [M][N]                   (final output)
// Block tile 128(M) x 64(N), K-step 32, 8 waves, wave tile 32x32 (2x2 WMMA).
// bf16-A path stages its tile with GLOBAL_LOAD_ASYNC_TO_LDS_B128 (ASYNCcnt).
// ---------------------------------------------------------------------------
template <typename TA, int MODE>
__global__ __launch_bounds__(256)
void gemm_wmma_kernel(const TA* __restrict__ A, const float* __restrict__ W,
                      const float* __restrict__ bias, void* __restrict__ Out,
                      int M, int N, int K, float oscale)
{
    __shared__ __bf16 As[128][32];
    __shared__ __bf16 Bs[64][32];

    const int tid    = threadIdx.x;
    const int lane   = tid & 31;
    const int wave   = tid >> 5;
    const int wm     = wave >> 1;         // 0..3  (M direction, 32 rows each)
    const int wn     = wave & 1;          // 0..1  (N direction, 32 cols each)
    const int lane16 = lane & 15;
    const int half   = lane >> 4;
    const int m0     = blockIdx.y * 128;
    const int n0     = blockIdx.x * 64;

    v8f c[2][2];
#pragma unroll
    for (int i = 0; i < 2; ++i)
#pragma unroll
        for (int j = 0; j < 2; ++j) c[i][j] = v8f_zero();

    const int arow = tid >> 1, akk = (tid & 1) * 16;   // A stage: 16 elems/thr
    const int brow = tid >> 2, bkk = (tid & 3) * 8;    // B stage:  8 elems/thr

    for (int k0 = 0; k0 < K; k0 += 32) {
        // ---- stage A tile (128x32) as bf16 ----
        {
            const TA* src = A + (size_t)(m0 + arow) * K + k0 + akk;
            __bf16*   dst = &As[arow][akk];
            if constexpr (sizeof(TA) == 4) {
                const float4* s4 = (const float4*)src;
#pragma unroll
                for (int q = 0; q < 4; ++q) {
                    float4 f = s4[q];
                    dst[q * 4 + 0] = (__bf16)f.x;
                    dst[q * 4 + 1] = (__bf16)f.y;
                    dst[q * 4 + 2] = (__bf16)f.z;
                    dst[q * 4 + 3] = (__bf16)f.w;
                }
            } else {
                // bf16 A: raw 32-byte copy, async global->LDS (no VGPR bounce).
                unsigned           ldso = (unsigned)(uintptr_t)dst;
                unsigned long long ga   = (unsigned long long)(uintptr_t)src;
                asm volatile(
                    "global_load_async_to_lds_b128 %0, %1, off\n\t"
                    "global_load_async_to_lds_b128 %0, %1, off offset:16"
                    :: "v"(ldso), "v"(ga) : "memory");
            }
            if (k0 + 32 < K)
                __builtin_prefetch(A + (size_t)(m0 + arow) * K + k0 + 32 + akk, 0, 0);
        }
        // ---- stage B tile (64x32) as bf16 ----
        {
            const float* src = W + (size_t)(n0 + brow) * K + k0 + bkk;
            __bf16*      dst = &Bs[brow][bkk];
            const float4* s4 = (const float4*)src;
#pragma unroll
            for (int q = 0; q < 2; ++q) {
                float4 f = s4[q];
                dst[q * 4 + 0] = (__bf16)f.x;
                dst[q * 4 + 1] = (__bf16)f.y;
                dst[q * 4 + 2] = (__bf16)f.z;
                dst[q * 4 + 3] = (__bf16)f.w;
            }
            if (k0 + 32 < K)
                __builtin_prefetch(W + (size_t)(n0 + brow) * K + k0 + 32 + bkk, 0, 0);
        }
        if constexpr (sizeof(TA) == 2)
            asm volatile("s_wait_asynccnt 0x0" ::: "memory");
        __syncthreads();

        // ---- fragments (per documented 16-bit WMMA VGPR layouts) ----
        v16bf afr[2], bfr[2];
#pragma unroll
        for (int i = 0; i < 2; ++i) {
            const int row = wm * 32 + i * 16 + lane16;
            const int kb  = half * 8;
            uv16 t;
            t.h[0] = *(const bf8*)&As[row][kb];        // K = kb .. kb+7
            t.h[1] = *(const bf8*)&As[row][kb + 16];   // K = kb+16 .. kb+23
            afr[i] = t.v;
        }
#pragma unroll
        for (int j = 0; j < 2; ++j) {
            const int nr = wn * 32 + j * 16 + lane16;
            uv16 t;
            t.h[0] = *(const bf8*)&Bs[nr][half * 8];       // K = half*8 .. half*8+7
            t.h[1] = *(const bf8*)&Bs[nr][half * 8 + 16];  // K = half*8+16 .. half*8+23
            bfr[j] = t.v;
        }
#pragma unroll
        for (int i = 0; i < 2; ++i)
#pragma unroll
            for (int j = 0; j < 2; ++j)
                c[i][j] = wmma_bf16(afr[i], bfr[j], c[i][j]);
        __syncthreads();
    }

    // ---- epilogue: bias + scale + store ----
#pragma unroll
    for (int i = 0; i < 2; ++i) {
#pragma unroll
        for (int j = 0; j < 2; ++j) {
            const int n_col = n0 + wn * 32 + j * 16 + lane16;
            const float bn  = bias[n_col];
#pragma unroll
            for (int r = 0; r < 8; ++r) {
                const int m_row = m0 + wm * 32 + i * 16 + half * 8 + r;
                const float val = (c[i][j][r] + bn) * oscale;
                if constexpr (MODE == 0) {
                    ((__bf16*)Out)[(size_t)m_row * N + n_col] = (__bf16)val;
                } else if constexpr (MODE == 1) {
                    const int bb = m_row >> 11, l = m_row & 2047;
                    const int hh = n_col >> 6,  d = n_col & 63;
                    ((__bf16*)Out)[(size_t)((bb * 16 + hh) * 64 + d) * 2048 + l] =
                        (__bf16)val;
                } else {
                    ((float*)Out)[(size_t)m_row * N + n_col] = val;
                }
            }
        }
    }
}

// ---------------------------------------------------------------------------
// Flash attention per (b, h).  Block = 4 waves, each wave owns 16 query rows.
//   Qb: bf16 [B*L][E] row-major, PRE-SCALED by 1/sqrt(E)=1/32.
//   Kb: bf16 [B*L][E] row-major.  Vt: bf16 [b][h][d][L].
//   Ob: bf16 [B*L][E] row-major (attention output, feeds final GEMM).
// Row max: single-instruction DPP butterfly. Row sum: WMMA against an
// all-ones B matrix, whose C-fragment row layout matches the O accumulator.
// ---------------------------------------------------------------------------
__global__ __launch_bounds__(128)
void attention_kernel(const __bf16* __restrict__ Qb, const __bf16* __restrict__ Kb,
                      const __bf16* __restrict__ Vt, __bf16* __restrict__ Ob)
{
    __shared__ __bf16 Pb[4][16][32];   // wave-private P transpose bounce

    const int qblk = blockIdx.x;       // 0..31
    const int h    = blockIdx.y;       // 0..15
    const int b    = blockIdx.z;       // 0..3
    const int lane = threadIdx.x & 31;
    const int wave = threadIdx.x >> 5;
    const int lane16 = lane & 15;
    const int half   = lane >> 4;
    const int q0     = qblk * 64 + wave * 16;

    // Q fragments: A-matrix 16x64 -> two 16x32 k-steps (loaded once).
    const __bf16* qrow = Qb + (size_t)(b * 2048 + q0 + lane16) * 1024 + h * 64;
    v16bf aq[2];
#pragma unroll
    for (int ks = 0; ks < 2; ++ks) {
        uv16 t;
        t.h[0] = *(const bf8*)(qrow + ks * 32 + half * 8);
        t.h[1] = *(const bf8*)(qrow + ks * 32 + half * 8 + 16);
        aq[ks] = t.v;
    }

    // all-ones B fragment for row-sum WMMA
    v16bf vones;
#pragma unroll
    for (int i = 0; i < 16; ++i) vones[i] = (__bf16)1.0f;

    v8f o[4];
#pragma unroll
    for (int dt = 0; dt < 4; ++dt) o[dt] = v8f_zero();
    v8f ls = v8f_zero();               // running row-sums (softmax denominator)
    float mrow[8];
#pragma unroll
    for (int r = 0; r < 8; ++r) mrow[r] = -1e30f;

    for (int kc = 0; kc < 2048; kc += 32) {
        // ---- S = Q @ K^T for 32 kv columns (two 16-col tiles) ----
        v8f s[2];
#pragma unroll
        for (int t = 0; t < 2; ++t) {
            const int kvcol = kc + t * 16 + lane16;
            const __bf16* krow = Kb + (size_t)(b * 2048 + kvcol) * 1024 + h * 64;
            v8f acc = v8f_zero();
#pragma unroll
            for (int ks = 0; ks < 2; ++ks) {
                uv16 bk;
                bk.h[0] = *(const bf8*)(krow + ks * 32 + half * 8);
                bk.h[1] = *(const bf8*)(krow + ks * 32 + half * 8 + 16);
                acc = wmma_bf16(aq[ks], bk.v, acc);
            }
            s[t] = acc;
        }

        // ---- online softmax (rows live within a 16-lane DPP row) ----
        float fac[8];
#pragma unroll
        for (int r = 0; r < 8; ++r) {
            const float e0 = s[0][r];        // Q pre-scaled: S already scaled
            const float e1 = s[1][r];
            const float mx   = red_max16(fmaxf(e0, e1));
            const float mnew = fmaxf(mrow[r], mx);
            const float f    = __expf(mrow[r] - mnew);
            const float p0   = __expf(e0 - mnew);
            const float p1   = __expf(e1 - mnew);
            mrow[r] = mnew;
            fac[r]  = f;
            Pb[wave][half * 8 + r][lane16]      = (__bf16)p0;
            Pb[wave][half * 8 + r][16 + lane16] = (__bf16)p1;
        }
#pragma unroll
        for (int dt = 0; dt < 4; ++dt)
#pragma unroll
            for (int r = 0; r < 8; ++r) o[dt][r] *= fac[r];
#pragma unroll
        for (int r = 0; r < 8; ++r) ls[r] *= fac[r];

        // wave-private LDS bounce: C-layout -> A-layout transpose of P.
        // (DS completes in-order within a wave, so only the RAW wait is needed.)
        asm volatile("s_wait_dscnt 0x0" ::: "memory");
        uv16 ap;
        ap.h[0] = *(const bf8*)&Pb[wave][lane16][half * 8];
        ap.h[1] = *(const bf8*)&Pb[wave][lane16][16 + half * 8];

        // row sums: ls += P @ ones  (row layout identical to O fragments)
        ls = wmma_bf16(ap.v, vones, ls);

        // ---- O += P @ V  (V^T layout gives contiguous per-lane loads) ----
#pragma unroll
        for (int dt = 0; dt < 4; ++dt) {
            const __bf16* vrow =
                Vt + (size_t)((b * 16 + h) * 64 + dt * 16 + lane16) * 2048;
            uv16 bv;
            bv.h[0] = *(const bf8*)(vrow + kc + half * 8);
            bv.h[1] = *(const bf8*)(vrow + kc + half * 8 + 16);
            o[dt] = wmma_bf16(ap.v, bv.v, o[dt]);
        }
    }

    // ---- normalize and store ----
    float inv[8];
#pragma unroll
    for (int r = 0; r < 8; ++r) inv[r] = 1.0f / ls[r];
#pragma unroll
    for (int dt = 0; dt < 4; ++dt) {
#pragma unroll
        for (int r = 0; r < 8; ++r) {
            const float val = o[dt][r] * inv[r];
            const int row = b * 2048 + q0 + half * 8 + r;
            const int col = h * 64 + dt * 16 + lane16;
            Ob[(size_t)row * 1024 + col] = (__bf16)val;
        }
    }
}

// ---------------------------------------------------------------------------
extern "C" void kernel_launch(void* const* d_in, const int* in_sizes, int n_in,
                              void* d_out, int out_size, void* d_ws, size_t ws_size,
                              hipStream_t stream)
{
    const float* query = (const float*)d_in[0];
    const float* key_  = (const float*)d_in[1];
    const float* value = (const float*)d_in[2];
    const float* Wq = (const float*)d_in[3];  const float* bq = (const float*)d_in[4];
    const float* Wk = (const float*)d_in[5];  const float* bk = (const float*)d_in[6];
    const float* Wv = (const float*)d_in[7];  const float* bv = (const float*)d_in[8];
    const float* Wo = (const float*)d_in[9];  const float* bo = (const float*)d_in[10];
    float* out = (float*)d_out;

    const int M = 4 * 2048;   // 8192 token rows
    const int E = 1024;
    const size_t np = (size_t)M * E;   // elems per bf16 tensor (16 MB each)

    __bf16* Qb = (__bf16*)d_ws;
    __bf16* Kb = Qb + np;
    __bf16* Vt = Kb + np;          // [b][h][d][L]
    __bf16* Ob = Vt + np;          // 4 * 16 MB = 64 MB workspace

    dim3 gg(E / 64, M / 128);      // (16, 64)

    const float qscale = 1.0f / 32.0f;   // 1/sqrt(E), folded into Q projection

    gemm_wmma_kernel<float, 0><<<gg, 256, 0, stream>>>(query, Wq, bq, Qb, M, E, E, qscale);
    gemm_wmma_kernel<float, 0><<<gg, 256, 0, stream>>>(key_,  Wk, bk, Kb, M, E, E, 1.0f);
    gemm_wmma_kernel<float, 1><<<gg, 256, 0, stream>>>(value, Wv, bv, Vt, M, E, E, 1.0f);

    attention_kernel<<<dim3(32, 16, 4), 128, 0, stream>>>(Qb, Kb, Vt, Ob);

    gemm_wmma_kernel<__bf16, 2><<<gg, 256, 0, stream>>>(Ob, Wo, bo, out, M, E, E, 1.0f);
}
